// ScoreModel_Time_UNetlike_edm_52810917871737
// MI455X (gfx1250) — matrix operand order repaired and checked by reading, not compiled, not run on hardware
//
#include <hip/hip_runtime.h>
#include <hip/hip_bf16.h>

// ---------------------------------------------------------------------------
// Problem constants
// ---------------------------------------------------------------------------
#define BATCH 16384
#define NDIM  128
#define NHID  2048
#define TEMB  128
#define NBLK  7

typedef __attribute__((ext_vector_type(16))) _Float16 v16h;
typedef __attribute__((ext_vector_type(8)))  _Float16 v8h;
typedef __attribute__((ext_vector_type(8)))  float    v8f;
typedef __attribute__((ext_vector_type(4)))  int      v4i;

// Async global->LDS support (CDNA5). Guarded so compile never hard-fails.
#if defined(__AMDGCN__) && __has_builtin(__builtin_amdgcn_global_load_async_to_lds_b128)
#define HAS_ASYNC_LDS 1
typedef __attribute__((address_space(1))) v4i glb_v4i;
typedef __attribute__((address_space(3))) v4i lds_v4i;
#else
#define HAS_ASYNC_LDS 0
#endif

static __device__ __forceinline__ void wait_async_done() {
#if HAS_ASYNC_LDS
#if __has_builtin(__builtin_amdgcn_s_wait_asynccnt)
    __builtin_amdgcn_s_wait_asynccnt(0);
#else
    asm volatile("s_wait_asynccnt 0" ::: "memory");
#endif
#endif
}

// ---------------------------------------------------------------------------
// f32 -> f16 conversion (weights, head input)
// ---------------------------------------------------------------------------
__global__ void cvt_f32_f16(const float* __restrict__ src, _Float16* __restrict__ dst, int n) {
    int i = blockIdx.x * 256 + threadIdx.x;
    if (i < n) dst[i] = (_Float16)src[i];
}

// ---------------------------------------------------------------------------
// Embedding + input scaling.
// ---------------------------------------------------------------------------
__global__ void embed_kernel(const float* __restrict__ x, const float* __restrict__ t,
                             const float* __restrict__ fW,
                             float* __restrict__ h0, _Float16* __restrict__ x16,
                             _Float16* __restrict__ emb16, float* __restrict__ stdv) {
    int b = blockIdx.x * 256 + threadIdx.x;
    if (b >= BATCH) return;
    const float LOG_SIGMA = 3.2188758248682006f;   // ln(25)
    float tt  = t[b];
    float stdb = sqrtf((expf(2.0f * tt * LOG_SIGMA) - 1.0f) / (2.0f * LOG_SIGMA));
    stdv[b] = stdb;
    float lnstd = logf(stdb) * 0.25f;
    #pragma unroll 4
    for (int j = 0; j < TEMB / 2; ++j) {
        float ph = lnstd * fW[j] * 6.283185307179586f;
        emb16[(size_t)b * TEMB + j]            = (_Float16)sinf(ph);
        emb16[(size_t)b * TEMB + TEMB / 2 + j] = (_Float16)cosf(ph);
    }
    float rs = rsqrtf(1.0f + stdb * stdb);
    #pragma unroll 4
    for (int c = 0; c < NDIM; ++c) {
        float v = x[(size_t)b * NDIM + c] * rs;
        h0[(size_t)b * NDIM + c]  = v;
        x16[(size_t)b * NDIM + c] = (_Float16)v;
    }
}

// ---------------------------------------------------------------------------
// Fused LayerNorm + SiLU, f32 in -> f16 out. One wave per row (wave32).
// ---------------------------------------------------------------------------
__global__ void ln_silu_kernel(const float* __restrict__ in, const float* __restrict__ w,
                               const float* __restrict__ b, _Float16* __restrict__ out, int fin) {
    int row  = blockIdx.x * 8 + (threadIdx.x >> 5);
    int lane = threadIdx.x & 31;
    const float* p = in + (size_t)row * fin;
    float s = 0.0f, sq = 0.0f;
    for (int c = lane; c < fin; c += 32) { float v = p[c]; s += v; sq += v * v; }
    #pragma unroll
    for (int off = 16; off; off >>= 1) { s += __shfl_xor(s, off, 32); sq += __shfl_xor(sq, off, 32); }
    float mean = s / (float)fin;
    float var  = sq / (float)fin - mean * mean;
    float r    = rsqrtf(var + 1e-5f);
    _Float16* q = out + (size_t)row * fin;
    for (int c = lane; c < fin; c += 32) {
        float y = (p[c] - mean) * r * w[c] + b[c];
        y = y / (1.0f + __expf(-y));
        q[c] = (_Float16)y;
    }
}

// ---------------------------------------------------------------------------
// Fused LayerNorm + affine modulation + SiLU.
// ---------------------------------------------------------------------------
__global__ void modulate_kernel(const float* __restrict__ tin, const float* __restrict__ aff,
                                const float* __restrict__ w, const float* __restrict__ b,
                                _Float16* __restrict__ out) {
    int row  = blockIdx.x * 8 + (threadIdx.x >> 5);
    int lane = threadIdx.x & 31;
    const float* p  = tin + (size_t)row * NHID;
    const float* ar = aff + (size_t)row * (2 * NHID);
    float s = 0.0f, sq = 0.0f;
    for (int c = lane; c < NHID; c += 32) { float v = p[c]; s += v; sq += v * v; }
    #pragma unroll
    for (int off = 16; off; off >>= 1) { s += __shfl_xor(s, off, 32); sq += __shfl_xor(sq, off, 32); }
    float mean = s / (float)NHID;
    float var  = sq / (float)NHID - mean * mean;
    float r    = rsqrtf(var + 1e-5f);
    _Float16* q = out + (size_t)row * NHID;
    for (int c = lane; c < NHID; c += 32) {
        float ln = (p[c] - mean) * r * w[c] + b[c];
        float y  = ar[NHID + c] + ln * (ar[c] + 1.0f);
        y = y / (1.0f + __expf(-y));
        q[c] = (_Float16)y;
    }
}

// ---------------------------------------------------------------------------
// WMMA GEMM:  out[M,N] = A16[M,K] * W16[N,K]^T + bias[N] (+ skip[M,N])
// headMode: out = (acc+bias)/std[m] - xorig[m,n]/(1+std^2)
// Block tile: 128 x BN_, 8 waves in WM_ x WN_ grid, each wave MFRAG*16 x NFRAG*16.
// BK=64, double-buffered LDS; global->LDS via async-to-LDS when available.
// ---------------------------------------------------------------------------
#define BM 128
#define BK 64
#define LDK 72   // padded row pitch in halves (144B)

static __device__ __forceinline__ v16h cat16(v8h lo, v8h hi) {
    v16h r;
    #pragma unroll
    for (int i = 0; i < 8; ++i) { r[i] = lo[i]; r[8 + i] = hi[i]; }
    return r;
}

template <int BN_, int WM_, int WN_, int MFRAG, int NFRAG>
__global__ __launch_bounds__(256)
void gemm_wmma_f16(const _Float16* __restrict__ A, const _Float16* __restrict__ W,
                   const float* __restrict__ bias, const float* __restrict__ skip,
                   float* __restrict__ out, int M, int N, int K,
                   int headMode, const float* __restrict__ stdv,
                   const float* __restrict__ xorig) {
    __shared__ _Float16 sA[2 * BM * LDK];
    __shared__ _Float16 sB[2 * BN_ * LDK];

    const int tid  = threadIdx.x;
    const int wid  = tid >> 5;
    const int lane = tid & 31;
    const int wm   = wid % WM_;
    const int wn   = wid / WM_;
    const int bm   = blockIdx.y * BM;
    const int bn   = blockIdx.x * BN_;

    v8f acc[MFRAG][NFRAG];
    #pragma unroll
    for (int mi = 0; mi < MFRAG; ++mi)
        #pragma unroll
        for (int ni = 0; ni < NFRAG; ++ni) {
            v8f z = {};
            acc[mi][ni] = z;
        }

    // Global -> LDS tile copy: (BM + BN_) rows x 64 halves, 8 halves per thread-step.
    auto loadTile = [&](int buf, int k0) {
        const _Float16* ga = A + (size_t)bm * K + k0;
        const _Float16* gw = W + (size_t)bn * K + k0;
        #pragma unroll
        for (int i = 0; i < (BM + BN_) * BK / 2048; ++i) {
            int idx = i * 256 + tid;
            int r   = idx >> 3;          // combined row 0 .. BM+BN_-1
            int c   = (idx & 7) << 3;    // 0..56 in 8-half steps
            const _Float16* g;
            _Float16* l;
            if (r < BM) {
                g = &ga[(size_t)r * K + c];
                l = &sA[buf * BM * LDK + r * LDK + c];
            } else {
                int rb = r - BM;
                g = &gw[(size_t)rb * K + c];
                l = &sB[buf * BN_ * LDK + rb * LDK + c];
            }
#if HAS_ASYNC_LDS
            __builtin_amdgcn_global_load_async_to_lds_b128(
                (glb_v4i*)g, (lds_v4i*)l, 0, 0);
#else
            *(v8h*)l = *(const v8h*)g;
#endif
        }
    };

    const int nk = K / BK;
    loadTile(0, 0);
    wait_async_done();
    __syncthreads();

    const int h8  = (lane >> 4) << 3;   // A-frag sub-offset: 0 or 8 halves
    const int g16 = (lane >> 4) << 3;   // B-frag sub-offset: 0 or 8 halves
    const int ml  = lane & 15;

    for (int kt = 0; kt < nk; ++kt) {
        const int cur = kt & 1;
        if (kt + 1 < nk) loadTile(cur ^ 1, (kt + 1) * BK);

        #pragma unroll
        for (int kk = 0; kk < BK; kk += 32) {
            v16h af[MFRAG];
            v16h bf[NFRAG];
            #pragma unroll
            for (int mi = 0; mi < MFRAG; ++mi) {
                int row = wm * (MFRAG * 16) + mi * 16 + ml;
                const _Float16* p = &sA[cur * BM * LDK + row * LDK + kk];
                v8h lo = *(const v8h*)&p[h8];          // K = kk + h8 .. +7
                v8h hi = *(const v8h*)&p[16 + h8];     // K = kk + 16 + h8 .. +7
                af[mi] = cat16(lo, hi);
            }
            #pragma unroll
            for (int ni = 0; ni < NFRAG; ++ni) {
                int col = wn * (NFRAG * 16) + ni * 16 + ml;
                const _Float16* p = &sB[cur * BN_ * LDK + col * LDK + kk + g16];
                v8h lo = *(const v8h*)&p[0];
                v8h hi = *(const v8h*)&p[16];
                bf[ni] = cat16(lo, hi);
            }
            #pragma unroll
            for (int mi = 0; mi < MFRAG; ++mi)
                #pragma unroll
                for (int ni = 0; ni < NFRAG; ++ni)
                    acc[mi][ni] = __builtin_amdgcn_wmma_f32_16x16x32_f16(
                        false, af[mi], false, bf[ni], (short)0, acc[mi][ni], false, false);
        }
        wait_async_done();
        __syncthreads();
    }

    // Epilogue.  C/D layout: lanes 0-15 -> M=v, lanes 16-31 -> M=8+v; N = lane&15.
    const int mofs = (lane >> 4) << 3;
    #pragma unroll
    for (int mi = 0; mi < MFRAG; ++mi) {
        #pragma unroll
        for (int ni = 0; ni < NFRAG; ++ni) {
            int col = bn + wn * (NFRAG * 16) + ni * 16 + ml;
            float bv = bias[col];
            #pragma unroll
            for (int v = 0; v < 8; ++v) {
                int row  = bm + wm * (MFRAG * 16) + mi * 16 + mofs + v;
                size_t o = (size_t)row * N + col;
                float val = acc[mi][ni][v] + bv;
                if (!headMode) {
                    if (skip) val += skip[o];
                    out[o] = val;
                } else {
                    float sd = stdv[row];
                    out[o] = val / sd - xorig[o] / (1.0f + sd * sd);
                }
            }
        }
    }
}

// Big-tile variant: 128x256 block, 2x4 waves, 64x64 per wave.
#define GEMM_BIG  gemm_wmma_f16<256, 2, 4, 4, 4>
// Small-N variant (head): 128x128 block, 4x2 waves, 32x64 per wave.
#define GEMM_SMALL gemm_wmma_f16<128, 4, 2, 2, 4>

// ---------------------------------------------------------------------------
// Host launcher
// ---------------------------------------------------------------------------
extern "C" void kernel_launch(void* const* d_in, const int* in_sizes, int n_in,
                              void* d_out, int out_size, void* d_ws, size_t ws_size,
                              hipStream_t stream) {
    (void)in_sizes; (void)n_in; (void)out_size; (void)ws_size;

    const float* x_in = (const float*)d_in[0];
    const float* t_in = (const float*)d_in[1];
    const float* fW   = (const float*)d_in[2];

    struct Blk {
        const float *n0w, *n0b, *fc0w, *fc0b, *affw, *affb, *n1w, *n1b, *fc1w, *fc1b;
        const float *skw, *skb;
        int fin;
    };
    Blk blk[NBLK];
    int idx = 3;
    for (int i = 0; i < NBLK; ++i) {
        blk[i].fin  = (i == 0) ? NDIM : NHID;
        blk[i].n0w  = (const float*)d_in[idx++];
        blk[i].n0b  = (const float*)d_in[idx++];
        blk[i].fc0w = (const float*)d_in[idx++];
        blk[i].fc0b = (const float*)d_in[idx++];
        blk[i].affw = (const float*)d_in[idx++];
        blk[i].affb = (const float*)d_in[idx++];
        blk[i].n1w  = (const float*)d_in[idx++];
        blk[i].n1b  = (const float*)d_in[idx++];
        blk[i].fc1w = (const float*)d_in[idx++];
        blk[i].fc1b = (const float*)d_in[idx++];
        if (i == 0) { blk[i].skw = (const float*)d_in[idx++]; blk[i].skb = (const float*)d_in[idx++]; }
        else        { blk[i].skw = nullptr; blk[i].skb = nullptr; }
    }
    const float* head_w = (const float*)d_in[idx++];
    const float* head_b = (const float*)d_in[idx++];

    // ---- carve workspace ----
    char* base = (char*)d_ws;
    size_t cur = 0;
    auto alloc = [&](size_t bytes) -> void* {
        void* p = base + cur;
        cur = (cur + bytes + 255) & ~(size_t)255;
        return p;
    };

    _Float16* fc0_16[NBLK];
    _Float16* aff_16[NBLK];
    _Float16* fc1_16[NBLK];
    for (int i = 0; i < NBLK; ++i) {
        fc0_16[i] = (_Float16*)alloc((size_t)NHID * blk[i].fin * 2);
        aff_16[i] = (_Float16*)alloc((size_t)2 * NHID * TEMB * 2);
        fc1_16[i] = (_Float16*)alloc((size_t)NHID * NHID * 2);
    }
    _Float16* skip16 = (_Float16*)alloc((size_t)NHID * NDIM * 2);
    _Float16* head16 = (_Float16*)alloc((size_t)NDIM * NHID * 2);
    _Float16* emb16  = (_Float16*)alloc((size_t)BATCH * TEMB * 2);
    _Float16* x16    = (_Float16*)alloc((size_t)BATCH * NDIM * 2);
    _Float16* gbuf16 = (_Float16*)alloc((size_t)BATCH * NHID * 2);
    float* hA   = (float*)alloc((size_t)BATCH * NHID * 4);
    float* hB   = (float*)alloc((size_t)BATCH * NHID * 4);
    float* tbuf = (float*)alloc((size_t)BATCH * NHID * 4);
    float* abuf = (float*)alloc((size_t)BATCH * 2 * NHID * 4);
    float* sbuf = (float*)alloc((size_t)BATCH * NHID * 4);
    float* stdv = (float*)alloc((size_t)BATCH * 4);

    auto cvt = [&](const float* s, _Float16* d, int n) {
        cvt_f32_f16<<<(n + 255) / 256, 256, 0, stream>>>(s, d, n);
    };

    // ---- weight conversion f32 -> f16 ----
    for (int i = 0; i < NBLK; ++i) {
        cvt(blk[i].fc0w, fc0_16[i], NHID * blk[i].fin);
        cvt(blk[i].affw, aff_16[i], 2 * NHID * TEMB);
        cvt(blk[i].fc1w, fc1_16[i], NHID * NHID);
    }
    cvt(blk[0].skw, skip16, NHID * NDIM);
    cvt(head_w, head16, NDIM * NHID);

    // ---- time embedding + input scaling ----
    embed_kernel<<<BATCH / 256, 256, 0, stream>>>(x_in, t_in, fW, hA, x16, emb16, stdv);

    // ---- blocks ----
    float* curIn  = hA;
    float* curOut = hB;
    for (int i = 0; i < NBLK; ++i) {
        const int fin = blk[i].fin;
        ln_silu_kernel<<<BATCH / 8, 256, 0, stream>>>(curIn, blk[i].n0w, blk[i].n0b, gbuf16, fin);
        GEMM_BIG<<<dim3(NHID / 256, BATCH / BM), 256, 0, stream>>>(
            gbuf16, fc0_16[i], blk[i].fc0b, nullptr, tbuf, BATCH, NHID, fin, 0, nullptr, nullptr);
        GEMM_BIG<<<dim3(2 * NHID / 256, BATCH / BM), 256, 0, stream>>>(
            emb16, aff_16[i], blk[i].affb, nullptr, abuf, BATCH, 2 * NHID, TEMB, 0, nullptr, nullptr);
        modulate_kernel<<<BATCH / 8, 256, 0, stream>>>(tbuf, abuf, blk[i].n1w, blk[i].n1b, gbuf16);
        const float* skipsrc;
        if (i == 0) {
            GEMM_BIG<<<dim3(NHID / 256, BATCH / BM), 256, 0, stream>>>(
                x16, skip16, blk[0].skb, nullptr, sbuf, BATCH, NHID, NDIM, 0, nullptr, nullptr);
            skipsrc = sbuf;
        } else {
            skipsrc = curIn;
        }
        GEMM_BIG<<<dim3(NHID / 256, BATCH / BM), 256, 0, stream>>>(
            gbuf16, fc1_16[i], blk[i].fc1b, skipsrc, curOut, BATCH, NHID, NHID, 0, nullptr, nullptr);
        float* tmp = curIn; curIn = curOut; curOut = tmp;
    }

    // ---- head: pred = (h @ head_w.T + head_b)/std - x/(1+std^2) ----
    cvt(curIn, gbuf16, BATCH * NHID);
    GEMM_SMALL<<<dim3(NDIM / 128, BATCH / BM), 256, 0, stream>>>(
        gbuf16, head16, head_b, nullptr, (float*)d_out, BATCH, NDIM, NHID, 1, stdv, x_in);
}
